// Attention_15418932592994
// MI455X (gfx1250) — compile-verified, matrix operand order checked by reading, not run on hardware
//
#include <hip/hip_runtime.h>
#include <hip/hip_bf16.h>

// ---------------------------------------------------------------------------
// Attention block for MI455X (gfx1250, wave32, WMMA + async-tensor engines).
// B=1, S=2048, D=4096, H=32, KVH=8, HD=128, causal, start_pos=0.
//
// All matmuls run on v_wmma_f32_16x16x32_f16 (f32 accumulate). GEMMs are
// block-tiled 128x128 through LDS with double-buffered async staging:
//   - A tile: Tensor Data Mover (tensor_load_to_lds, TENSORcnt).
//   - B tile: global_load_async_to_lds_b128 (ASYNCcnt) inline asm.
// sched_barrier(0) separates the fragment-load group from the WMMA group so
// the loads form one clause and WMMAs pipeline on partial dscnt waits.
// __launch_bounds__(256,1) gives the allocator the full VGPR budget so the
// widened live ranges do not spill to scratch.
// ---------------------------------------------------------------------------

typedef _Float16 f16;
typedef __attribute__((ext_vector_type(8)))  _Float16 v8h;
typedef __attribute__((ext_vector_type(16))) _Float16 v16h;
typedef __attribute__((ext_vector_type(8)))  float    v8f;
typedef unsigned int u32x4 __attribute__((ext_vector_type(4)));
typedef int          i32x4 __attribute__((ext_vector_type(4)));
typedef int          i32x8 __attribute__((ext_vector_type(8)));

#define S_LEN   2048
#define D_MODEL 4096
#define NH      32
#define NKVH    8
#define HD      128
#define NREP    4
#define KVD     (NKVH * HD)   // 1024
#define KSTEP   32            // k-depth per staged tile

__device__ __forceinline__ void sched_fence() {
#if __has_builtin(__builtin_amdgcn_sched_barrier)
  __builtin_amdgcn_sched_barrier(0);   // nothing may be reordered across this
#endif
}

// Generic pointer -> 32-bit LDS byte offset (LDS aperture keeps the offset in
// the low 32 bits of the flat address; hardware uses ADDR[31:0] for LDS).
__device__ __forceinline__ unsigned ldsoff(const void* p) {
  return (unsigned)(unsigned long long)(uintptr_t)p;
}

// ASYNCcnt-tracked 16-byte global -> LDS copy (CDNA5 async data mover path).
__device__ __forceinline__ void async_b128(unsigned loff, const f16* g) {
  asm volatile("global_load_async_to_lds_b128 %0, %1, off"
               :: "v"(loff), "v"(g) : "memory");
}
__device__ __forceinline__ void wait_asynccnt0() {
  asm volatile("s_wait_asynccnt 0x0" ::: "memory");
}

#if __has_builtin(__builtin_amdgcn_tensor_load_to_lds)
// TDM: load one 128-row x 32-f16 tile (rows of 16 dwords) global -> LDS.
// Descriptor per CDNA5 ISA D# groups 0/1 (2D tensor).
__device__ __forceinline__ void tdm_load_tile_128x32(unsigned lds,
                                                     const f16* gaddr,
                                                     int k_ld_dw /* dwords/row */) {
  unsigned long long ga = (unsigned long long)(uintptr_t)gaddr;
  u32x4 g0;
  g0[0] = 1u;                                        // count=1, user descriptor
  g0[1] = lds;                                       // lds_addr
  g0[2] = (unsigned)(ga & 0xffffffffu);              // global_addr[31:0]
  g0[3] = (unsigned)((ga >> 32) & 0x01ffffffu)       // global_addr[56:32]
          | (2u << 30);                              // type = 2 ("image")
  i32x8 g1;
  g1[0] = (int)(2u << 16);                           // data_size = 4B units
  g1[1] = (int)(((unsigned)k_ld_dw & 0xffffu) << 16);        // tensor_dim0 lo16
  g1[2] = (int)(((unsigned)k_ld_dw >> 16) | (128u << 16));   // dim0 hi | dim1 lo
  g1[3] = (int)(16u << 16);                          // dim1 hi=0 | tile_dim0=16 dw
  g1[4] = 128;                                       // tile_dim1=128, tile_dim2=0
  g1[5] = k_ld_dw;                                   // tensor_dim0_stride lo32
  g1[6] = 0;                                         // stride0 hi | stride1 lo
  g1[7] = 0;
  i32x4 z4 = {0, 0, 0, 0};
#if defined(__clang_major__) && (__clang_major__ >= 23)
  i32x8 z8 = {0, 0, 0, 0, 0, 0, 0, 0};
  __builtin_amdgcn_tensor_load_to_lds(g0, g1, z4, z4, z8, 0);
#else
  __builtin_amdgcn_tensor_load_to_lds(g0, g1, z4, z4, 0);
#endif
}
#endif

// Load a 16x32 f16 WMMA operand fragment from a row-major [rows][ldk] matrix
// in global memory (A-style / transposed-B-style lane layout, CDNA5 ISA).
__device__ __forceinline__ v16h load_frag16(const f16* __restrict__ base, int ldk,
                                            int row, int kb) {
  union { v16h v; v8h h[2]; } u;
  const f16* p = base + (size_t)row * (size_t)ldk + kb;
  u.h[0] = *(const v8h*)(p);
  u.h[1] = *(const v8h*)(p + 16);
  return u.v;
}

// Same fragment gather but from an LDS tile with row stride KSTEP.
__device__ __forceinline__ v16h lds_frag16(const f16* tile, int row, int kb) {
  union { v16h v; v8h h[2]; } u;
  const f16* p = tile + row * KSTEP + kb;
  u.h[0] = *(const v8h*)(p);
  u.h[1] = *(const v8h*)(p + 16);
  return u.v;
}

// ---------------------------------------------------------------------------
// Elementwise conversion / transposition passes (one-time, bandwidth-bound)
// ---------------------------------------------------------------------------
__global__ void cvt_f32_to_f16(const float* __restrict__ in, f16* __restrict__ out) {
  int idx = blockIdx.x * 256 + threadIdx.x;
  out[idx] = (f16)in[idx];
}

// w[K][N] f32  ->  wt[N][K] f16   (so WMMA B-operand loads contiguous K)
__global__ void transpose_to_f16(const float* __restrict__ w, f16* __restrict__ wt,
                                 int K, int N) {
  int o = blockIdx.x * 256 + threadIdx.x;   // index into wt
  int n = o / K;
  int k = o - n * K;
  wt[(size_t)o] = (f16)w[(size_t)k * N + n];
}

// ---------------------------------------------------------------------------
// Block-tiled WMMA GEMM:  C[M][N] f32 = A[M][K] f16 x Bt[N][K] f16
// 256-thread block (8 waves) computes a 128x128 C tile; k in steps of 32.
// A/B tiles double-buffered in LDS via TDM + async-to-LDS copies.
// Wave w computes rows [w*16, w*16+16) x all 128 cols (8 WMMA accumulators).
// Grid: (M/128)*(N/128) blocks.
// ---------------------------------------------------------------------------
__global__ void __launch_bounds__(256, 1)
gemm_f16_f32(const f16* __restrict__ A, const f16* __restrict__ Bt,
             float* __restrict__ C, int M, int N, int K) {
  __shared__ __align__(16) f16 Abuf[2][128 * KSTEP];   // 2 x 8 KB
  __shared__ __align__(16) f16 Bbuf[2][128 * KSTEP];   // 2 x 8 KB
  const int tid  = threadIdx.x;
  const int lane = tid & 31;
  const int wave = tid >> 5;
  const int lr   = lane & 15;
  const int hi   = lane >> 4;
  const int tilesN = N >> 7;
  const int tm = blockIdx.x / tilesN;
  const int tn = blockIdx.x - tm * tilesN;
  const f16* Ab = A  + (size_t)(tm * 128) * K;
  const f16* Bb = Bt + (size_t)(tn * 128) * K;

  // Stage one 128x32 tile with 512 async b128 copies spread over 256 threads.
  auto stage_async = [&](f16* lbase, const f16* gbase, int k0) {
#pragma unroll
    for (int i = 0; i < 2; ++i) {
      int lin = tid * 2 + i;              // 0..511
      int row = lin >> 2;
      int seg = (lin & 3) * 8;            // 8 f16 = 16 B
      async_b128(ldsoff(lbase + row * KSTEP + seg),
                 gbase + (size_t)row * K + k0 + seg);
    }
  };
  auto stage_a = [&](int buf, int k0) {
#if __has_builtin(__builtin_amdgcn_tensor_load_to_lds)
    if (wave == 0) tdm_load_tile_128x32(ldsoff(&Abuf[buf][0]), Ab + k0, K >> 1);
#else
    stage_async(&Abuf[buf][0], Ab, k0);
#endif
  };

  v8f acc[8] = {};
  const int ktiles = K / KSTEP;
  stage_a(0, 0);
  stage_async(&Bbuf[0][0], Bb, 0);

  for (int kt = 0; kt < ktiles; ++kt) {
    const int buf = kt & 1;
#if __has_builtin(__builtin_amdgcn_tensor_load_to_lds)
    if (wave == 0) __builtin_amdgcn_s_wait_tensorcnt((short)0);
#endif
    wait_asynccnt0();
    __syncthreads();                      // staged tile visible to all waves
    if (kt + 1 < ktiles) {                // overlap next DMA with compute
      stage_a(buf ^ 1, (kt + 1) * KSTEP);
      stage_async(&Bbuf[buf ^ 1][0], Bb, (kt + 1) * KSTEP);
    }
    // Preload ALL fragments (one clause, distinct regs, partial dscnt waits)
    v16h a = lds_frag16(&Abuf[buf][0], wave * 16 + lr, hi * 8);
    v16h bf[8];
#pragma unroll
    for (int s = 0; s < 8; ++s)
      bf[s] = lds_frag16(&Bbuf[buf][0], s * 16 + lr, hi * 8);
    sched_fence();   // ... then 8 WMMAs back-to-back; loads cannot sink past
#pragma unroll
    for (int s = 0; s < 8; ++s)
      acc[s] = __builtin_amdgcn_wmma_f32_16x16x32_f16(
          false, a, false, bf[s], (short)0, acc[s], false, false);
    sched_fence();
  }
  // C layout (ISA): lane = column lr, VGPR r = row r (lanes 0-15) / r+8 (16-31)
#pragma unroll
  for (int s = 0; s < 8; ++s) {
#pragma unroll
    for (int r = 0; r < 8; ++r) {
      int row = tm * 128 + wave * 16 + r + hi * 8;
      int col = tn * 128 + s * 16 + lr;
      C[(size_t)row * N + col] = acc[s][r];
    }
  }
}

// ---------------------------------------------------------------------------
// RoPE + layout kernels
// ---------------------------------------------------------------------------
__global__ void rope_q_kernel(const float* __restrict__ Q,
                              const float* __restrict__ cosT,
                              const float* __restrict__ sinT,
                              f16* __restrict__ Qr) {
  int idx = blockIdx.x * 256 + threadIdx.x;      // S*H*64
  int i = idx & 63;
  int h = (idx >> 6) & (NH - 1);
  int s = idx >> 11;
  float c  = cosT[s * 64 + i];
  float sn = sinT[s * 64 + i];
  size_t base = (size_t)s * D_MODEL + h * HD + 2 * i;
  float v0 = Q[base], v1 = Q[base + 1];
  Qr[base]     = (f16)(v0 * c - v1 * sn);
  Qr[base + 1] = (f16)(v0 * sn + v1 * c);
}

// K f32 (S, KVH*HD) -> Kr f16 (KVH, S, HD)  (rope applied; key-major rows)
__global__ void rope_k_kernel(const float* __restrict__ K,
                              const float* __restrict__ cosT,
                              const float* __restrict__ sinT,
                              f16* __restrict__ Kr) {
  int idx = blockIdx.x * 256 + threadIdx.x;      // S*KVH*64
  int i = idx & 63;
  int h = (idx >> 6) & (NKVH - 1);
  int s = idx >> 9;
  float c  = cosT[s * 64 + i];
  float sn = sinT[s * 64 + i];
  size_t src = (size_t)s * KVD + h * HD + 2 * i;
  float v0 = K[src], v1 = K[src + 1];
  size_t dst = ((size_t)h * S_LEN + s) * HD + 2 * i;
  Kr[dst]     = (f16)(v0 * c - v1 * sn);
  Kr[dst + 1] = (f16)(v0 * sn + v1 * c);
}

// V f32 (S, KVH*HD) -> Vt f16 (KVH, HD, S)  (hd-major so PV B-frags load contig keys)
__global__ void vt_kernel(const float* __restrict__ V, f16* __restrict__ Vt) {
  int idx = blockIdx.x * 256 + threadIdx.x;      // S*KVD
  int j = idx & (KVD - 1);
  int s = idx >> 10;
  int h  = j >> 7;
  int hd = j & (HD - 1);
  Vt[((size_t)h * HD + hd) * S_LEN + s] = (f16)V[(size_t)s * KVD + j];
}

// ---------------------------------------------------------------------------
// Flash attention: one wave per (head, 16-query tile). 32-key blocks.
// Qr: (S, H*HD) f16;  Kr: (KVH, S, HD) f16;  Vt: (KVH, HD, S) f16
// O: (S, H*HD) f16 attention output (pre-projection).
// ---------------------------------------------------------------------------
__global__ void __launch_bounds__(256, 1)
flash_attn_kernel(const f16* __restrict__ Qr,
                  const f16* __restrict__ Kr,
                  const f16* __restrict__ Vt,
                  f16* __restrict__ O) {
  __shared__ float pstage[8][16 * 32];           // per-wave P staging (16 KB)
  const int lane = threadIdx.x & 31;
  const int wave = threadIdx.x >> 5;
  const int lr   = lane & 15;
  const int hi   = lane >> 4;
  float* pbuf = pstage[wave];

  const int tile = blockIdx.x * 8 + wave;        // H * (S/16) = 4096 tiles
  const int qt  = tile & (S_LEN / 16 - 1);
  const int h   = tile >> 7;
  const int kvh = h >> 2;                        // NREP = 4
  const int q0  = qt * 16;

  // Q fragments: 16 x 128 (4 k-steps of 32)
  v16h qf[4];
#pragma unroll
  for (int s = 0; s < 4; ++s)
    qf[s] = load_frag16(Qr + h * HD, D_MODEL, q0 + lr, s * 32 + hi * 8);

  v8f  o[8] = {};                                // 16 x 128 f32 accumulator
  float m[8], l[8];
#pragma unroll
  for (int r = 0; r < 8; ++r) { m[r] = -INFINITY; l[r] = 0.0f; }

  const float scale = 0.08838834764831845f;      // 1/sqrt(128)
  const f16* Kh = Kr + (size_t)kvh * S_LEN * HD;
  const f16* Vh = Vt + (size_t)kvh * HD * S_LEN;

  for (int k0 = 0; k0 < q0 + 16; k0 += 32) {
    // ---- preload all 8 K fragments, then 8 WMMAs (S = Q K^T, 16x32) ----
    v16h kf[8];
#pragma unroll
    for (int s = 0; s < 4; ++s)
#pragma unroll
      for (int n = 0; n < 2; ++n)
        kf[s * 2 + n] = load_frag16(Kh, HD, k0 + n * 16 + lr, s * 32 + hi * 8);
    sched_fence();                               // loads form one clause
    v8f sf[2] = {};
#pragma unroll
    for (int s = 0; s < 4; ++s)
#pragma unroll
      for (int n = 0; n < 2; ++n)
        sf[n] = __builtin_amdgcn_wmma_f32_16x16x32_f16(
            false, qf[s], false, kf[s * 2 + n], (short)0, sf[n], false, false);

    // ---- preload V fragments early (overlap with softmax VALU work) ----
    v16h vf[8];
#pragma unroll
    for (int n = 0; n < 8; ++n)
      vf[n] = load_frag16(Vh, S_LEN, n * 16 + lr, k0 + hi * 8);

    // ---- causal mask + online softmax (per row r; row = q0 + r + hi*8) ----
#pragma unroll
    for (int r = 0; r < 8; ++r) {
      int qrow = q0 + r + hi * 8;
      float x0 = sf[0][r] * scale;
      float x1 = sf[1][r] * scale;
      if (k0 + lr      > qrow) x0 = -INFINITY;
      if (k0 + 16 + lr > qrow) x1 = -INFINITY;
      float mx = fmaxf(x0, x1);
#pragma unroll
      for (int d = 1; d < 16; d <<= 1) mx = fmaxf(mx, __shfl_xor(mx, d, 16));
      float mnew  = fmaxf(m[r], mx);
      float alpha = __expf(m[r] - mnew);
      float p0 = __expf(x0 - mnew);
      float p1 = __expf(x1 - mnew);
      float ps = p0 + p1;
#pragma unroll
      for (int d = 1; d < 16; d <<= 1) ps += __shfl_xor(ps, d, 16);
      l[r] = l[r] * alpha + ps;
      m[r] = mnew;
#pragma unroll
      for (int n = 0; n < 8; ++n) o[n][r] *= alpha;  // rescale accumulators
      // stage P row-major 16x32 in LDS for C-frag -> A-frag relayout
      pbuf[(r + hi * 8) * 32 + lr]      = p0;
      pbuf[(r + hi * 8) * 32 + 16 + lr] = p1;
    }
    // ---- read P back as a 16x32 f16 A-fragment ----
    union { v16h v; f16 e[16]; } pa;
#pragma unroll
    for (int j = 0; j < 8; ++j) {
      pa.e[j]     = (f16)pbuf[lr * 32 + hi * 8 + j];
      pa.e[8 + j] = (f16)pbuf[lr * 32 + 16 + hi * 8 + j];
    }
    sched_fence();
    // ---- O += P V : 8 WMMAs back-to-back on preloaded V fragments ----
#pragma unroll
    for (int n = 0; n < 8; ++n)
      o[n] = __builtin_amdgcn_wmma_f32_16x16x32_f16(
          false, pa.v, false, vf[n], (short)0, o[n], false, false);
    sched_fence();
  }
  // ---- normalize + store f16 (S, H*HD) ----
#pragma unroll
  for (int n = 0; n < 8; ++n) {
#pragma unroll
    for (int r = 0; r < 8; ++r) {
      int row = q0 + r + hi * 8;
      int col = h * HD + n * 16 + lr;
      O[(size_t)row * D_MODEL + col] = (f16)(o[n][r] / l[r]);
    }
  }
}

// ---------------------------------------------------------------------------
// Host-side orchestration
// ---------------------------------------------------------------------------
extern "C" void kernel_launch(void* const* d_in, const int* in_sizes, int n_in,
                              void* d_out, int out_size, void* d_ws, size_t ws_size,
                              hipStream_t stream) {
  (void)in_sizes; (void)n_in; (void)out_size; (void)ws_size;
  const float* x    = (const float*)d_in[0];
  const float* wq   = (const float*)d_in[1];
  const float* wk   = (const float*)d_in[2];
  const float* wv   = (const float*)d_in[3];
  const float* wo   = (const float*)d_in[4];
  const float* fcos = (const float*)d_in[5];
  const float* fsin = (const float*)d_in[6];
  // d_in[7] = mask (causal, reimplemented analytically), d_in[8] = start_pos (0)
  float* out = (float*)d_out;

  char* ws = (char*)d_ws;
  size_t off = 0;
  auto alloc = [&](size_t bytes) -> void* {
    void* p = ws + off;
    off += (bytes + 255) & ~(size_t)255;
    return p;
  };
  f16*   x16 = (f16*)  alloc((size_t)S_LEN * D_MODEL * 2);
  f16*   wqt = (f16*)  alloc((size_t)D_MODEL * D_MODEL * 2);
  f16*   wkt = (f16*)  alloc((size_t)KVD * D_MODEL * 2);
  f16*   wvt = (f16*)  alloc((size_t)KVD * D_MODEL * 2);
  f16*   wot = (f16*)  alloc((size_t)D_MODEL * D_MODEL * 2);
  float* Qf  = (float*)alloc((size_t)S_LEN * D_MODEL * 4);
  float* Kf  = (float*)alloc((size_t)S_LEN * KVD * 4);
  float* Vf  = (float*)alloc((size_t)S_LEN * KVD * 4);
  f16*   Qr  = (f16*)  alloc((size_t)S_LEN * D_MODEL * 2);
  f16*   Kr  = (f16*)  alloc((size_t)NKVH * S_LEN * HD * 2);
  f16*   Vt  = (f16*)  alloc((size_t)NKVH * HD * S_LEN * 2);
  f16*   AO  = (f16*)  alloc((size_t)S_LEN * D_MODEL * 2);

  // 1) one-time f16 conversion / transposition
  cvt_f32_to_f16<<<(S_LEN * D_MODEL) / 256, 256, 0, stream>>>(x, x16);
  transpose_to_f16<<<(D_MODEL * D_MODEL) / 256, 256, 0, stream>>>(wq, wqt, D_MODEL, D_MODEL);
  transpose_to_f16<<<(D_MODEL * KVD)     / 256, 256, 0, stream>>>(wk, wkt, D_MODEL, KVD);
  transpose_to_f16<<<(D_MODEL * KVD)     / 256, 256, 0, stream>>>(wv, wvt, D_MODEL, KVD);
  transpose_to_f16<<<(D_MODEL * D_MODEL) / 256, 256, 0, stream>>>(wo, wot, D_MODEL, D_MODEL);

  // 2) QKV projections on WMMA (block-tiled, async/TDM staged)
  gemm_f16_f32<<<(S_LEN / 128) * (D_MODEL / 128), 256, 0, stream>>>(
      x16, wqt, Qf, S_LEN, D_MODEL, D_MODEL);
  gemm_f16_f32<<<(S_LEN / 128) * (KVD / 128), 256, 0, stream>>>(
      x16, wkt, Kf, S_LEN, KVD, D_MODEL);
  gemm_f16_f32<<<(S_LEN / 128) * (KVD / 128), 256, 0, stream>>>(
      x16, wvt, Vf, S_LEN, KVD, D_MODEL);

  // 3) RoPE + attention layouts
  rope_q_kernel<<<(S_LEN * NH * 64)   / 256, 256, 0, stream>>>(Qf, fcos, fsin, Qr);
  rope_k_kernel<<<(S_LEN * NKVH * 64) / 256, 256, 0, stream>>>(Kf, fcos, fsin, Kr);
  vt_kernel<<<(S_LEN * KVD) / 256, 256, 0, stream>>>(Vf, Vt);

  // 4) flash attention (one wave per head x 16-query tile)
  flash_attn_kernel<<<(NH * (S_LEN / 16)) / 8, 256, 0, stream>>>(Qr, Kr, Vt, AO);

  // 5) output projection on WMMA -> f32 result
  gemm_f16_f32<<<(S_LEN / 128) * (D_MODEL / 128), 256, 0, stream>>>(
      AO, wot, out, S_LEN, D_MODEL, D_MODEL);
}
